// MambaBlock_20504173871376
// MI455X (gfx1250) — compile-verified, matrix operand order checked by reading, not run on hardware
//
#include <hip/hip_runtime.h>
#include <hip/hip_bf16.h>

typedef __attribute__((ext_vector_type(16))) _Float16 v16h;
typedef __attribute__((ext_vector_type(8)))  _Float16 v8h;
typedef __attribute__((ext_vector_type(8)))  float    v8f;

#define BATCH   4
#define SEQ     2048
#define ROWS    (BATCH * SEQ)      // 8192
#define DM      1024               // d_model
#define DI      2048               // d_inner
#define XZN     (2 * DI)           // 4096
#define DTR     64                 // dt_rank
#define DS      16                 // d_state
#define XPN     (DTR + 2 * DS)     // 96

// ---------------------------------------------------------------------------
// f32 -> f16 convert (grid-stride)
__global__ void cvt_f32_f16(const float* __restrict__ src, _Float16* __restrict__ dst, size_t n) {
    size_t i = (size_t)blockIdx.x * blockDim.x + threadIdx.x;
    size_t stride = (size_t)gridDim.x * blockDim.x;
    for (; i < n; i += stride) dst[i] = (_Float16)src[i];
}

// strided slice convert: dst[r*cols+c] = (f16) src[r*lds+c]
__global__ void cvt_slice_f16(const float* __restrict__ src, _Float16* __restrict__ dst,
                              int rows, int cols, int lds) {
    size_t total = (size_t)rows * cols;
    size_t i = (size_t)blockIdx.x * blockDim.x + threadIdx.x;
    size_t stride = (size_t)gridDim.x * blockDim.x;
    for (; i < total; i += stride) {
        int r = (int)(i / cols), c = (int)(i % cols);
        dst[i] = (_Float16)src[(size_t)r * lds + c];
    }
}

// ---------------------------------------------------------------------------
// residual add + RMSNorm.  One block per row (256 threads x 4 elems).
__global__ void addnorm_kernel(const float* __restrict__ hs, const float* __restrict__ res,
                               const float* __restrict__ w,
                               float* __restrict__ resid_out, _Float16* __restrict__ hh) {
    __shared__ float sm[256];
    const int row = blockIdx.x;
    const int tid = threadIdx.x;
    const float4 a = ((const float4*)(hs  + (size_t)row * DM))[tid];
    const float4 b = ((const float4*)(res + (size_t)row * DM))[tid];
    float4 rv = {a.x + b.x, a.y + b.y, a.z + b.z, a.w + b.w};
    sm[tid] = rv.x * rv.x + rv.y * rv.y + rv.z * rv.z + rv.w * rv.w;
    __syncthreads();
    for (int s = 128; s > 0; s >>= 1) {
        if (tid < s) sm[tid] += sm[tid + s];
        __syncthreads();
    }
    const float inv = rsqrtf(sm[0] * (1.0f / DM) + 1e-5f);
    const float4 w4 = ((const float4*)w)[tid];
    ((float4*)(resid_out + (size_t)row * DM))[tid] = rv;
    _Float16* hp = hh + (size_t)row * DM + tid * 4;
    hp[0] = (_Float16)(rv.x * inv * w4.x);
    hp[1] = (_Float16)(rv.y * inv * w4.y);
    hp[2] = (_Float16)(rv.z * inv * w4.z);
    hp[3] = (_Float16)(rv.w * inv * w4.w);
}

// ---------------------------------------------------------------------------
// WMMA NT GEMM: C[M,N] = A[M,K] * B[N,K]^T   (both K-contiguous, f16 in, f32 out)
// One wave computes a 16 x (16*NB) tile.  MODE 0: plain store; MODE 1: softplus(x+bias).
template <int NB, int MODE>
__global__ void gemm_nt_wmma(const _Float16* __restrict__ A, const _Float16* __restrict__ Bw,
                             float* __restrict__ C, const float* __restrict__ bias,
                             int M, int N, int K, int lda, int ldb, int ldc) {
    const int lane = threadIdx.x & 31;
    const int wave = threadIdx.x >> 5;
    const int ntn  = N / (16 * NB);
    const int tile = blockIdx.x * (blockDim.x >> 5) + wave;
    if (tile >= (M / 16) * ntn) return;
    const int tm = tile / ntn, tn = tile % ntn;
    const int m  = lane & 15;
    const int hf = lane >> 4;           // which half-wave: selects K sub-blocks

    v8f acc[NB];
    for (int j = 0; j < NB; ++j) {
        v8f z = {0.f, 0.f, 0.f, 0.f, 0.f, 0.f, 0.f, 0.f};
        acc[j] = z;
    }

    const _Float16* arow = A + (size_t)(tm * 16 + m) * lda;
    for (int k0 = 0; k0 < K; k0 += 32) {
        // A fragment: 16x32 f16.  lane half hf holds K = 8*hf + [0..7] and 16+8*hf + [0..7]
        const _Float16* ap = arow + k0 + 8 * hf;
        v16h av;
        {
            v8h lo = *(const v8h*)ap;
            v8h hi = *(const v8h*)(ap + 16);
            for (int i = 0; i < 8; ++i) { av[i] = lo[i]; av[i + 8] = hi[i]; }
        }
        for (int j = 0; j < NB; ++j) {
            const _Float16* bp =
                Bw + (size_t)(tn * 16 * NB + j * 16 + m) * ldb + k0 + 8 * hf;
            v16h bv;
            {
                v8h lo = *(const v8h*)bp;
                v8h hi = *(const v8h*)(bp + 16);
                for (int i = 0; i < 8; ++i) { bv[i] = lo[i]; bv[i + 8] = hi[i]; }
            }
            acc[j] = __builtin_amdgcn_wmma_f32_16x16x32_f16(
                false, av, false, bv, (short)0, acc[j], false, false);
        }
    }
    // D layout: VGPR i -> row (tm*16 + i + 8*hf), col (tn*16*NB + j*16 + (lane&15))
    for (int j = 0; j < NB; ++j) {
        const int col = tn * 16 * NB + j * 16 + m;
        float bs = (MODE == 1) ? bias[col] : 0.f;
        for (int i = 0; i < 8; ++i) {
            const int row = tm * 16 + i + 8 * hf;
            float v = acc[j][i];
            if (MODE == 1) {
                v += bs;
                v = (v > 20.f) ? v : log1pf(__expf(v));   // softplus
            }
            C[(size_t)row * ldc + col] = v;
        }
    }
}

// ---------------------------------------------------------------------------
// depthwise causal conv (width 4) + bias + SiLU on the x-half of xz.
// grid: (ROWS, DI/256); block 256.
__global__ void conv_silu_kernel(const float* __restrict__ xz, const float* __restrict__ cw,
                                 const float* __restrict__ cb,
                                 float* __restrict__ xf, _Float16* __restrict__ xh) {
    const int d = blockIdx.y * 256 + threadIdx.x;
    const int r = blockIdx.x;
    const int b = r >> 11, l = r & (SEQ - 1);
    float acc = cb[d];
    #pragma unroll
    for (int j = 0; j < 4; ++j) {
        int ll = l - 3 + j;
        if (ll >= 0)
            acc += cw[d * 4 + j] * xz[((size_t)(b * SEQ + ll)) * XZN + d];
    }
    const float s = acc / (1.f + __expf(-acc));      // SiLU
    xf[(size_t)r * DI + d] = s;
    xh[(size_t)r * DI + d] = (_Float16)s;
}

// ---------------------------------------------------------------------------
// selective scan: 16 lanes (one per state n) cooperate on one (b,d) scan.
// Fused: y = scan + x*D, then y *= SiLU(z); emit f16 for the out-proj GEMM.
__global__ void scan_kernel(const float* __restrict__ dt, const float* __restrict__ xf,
                            const float* __restrict__ xdbl, const float* __restrict__ xz,
                            const float* __restrict__ A_log, const float* __restrict__ Dp,
                            _Float16* __restrict__ yh) {
    const int gid = (int)((blockIdx.x * blockDim.x + threadIdx.x) >> 4);
    const int n   = threadIdx.x & 15;
    if (gid >= BATCH * DI) return;
    const int b = gid >> 11;            // DI == 2048
    const int d = gid & (DI - 1);
    const float Av = -__expf(A_log[d * DS + n]);
    const float Dv = Dp[d];
    float st = 0.f;
    for (int l = 0; l < SEQ; ++l) {
        const size_t r = (size_t)b * SEQ + l;
        const float dtv = dt[r * DI + d];
        const float xv  = xf[r * DI + d];
        const float Bv  = xdbl[r * XPN + DTR + n];
        const float Cv  = xdbl[r * XPN + DTR + DS + n];
        st = st * __expf(dtv * Av) + dtv * Bv * xv;
        float p = st * Cv;
        p += __shfl_xor(p, 1, 16);
        p += __shfl_xor(p, 2, 16);
        p += __shfl_xor(p, 4, 16);
        p += __shfl_xor(p, 8, 16);
        if (n == 0) {
            float y  = p + xv * Dv;
            float zv = xz[r * XZN + DI + d];
            y *= zv / (1.f + __expf(-zv));           // * SiLU(z)
            yh[r * DI + d] = (_Float16)y;
        }
    }
}

// ---------------------------------------------------------------------------
extern "C" void kernel_launch(void* const* d_in, const int* in_sizes, int n_in,
                              void* d_out, int out_size, void* d_ws, size_t ws_size,
                              hipStream_t stream) {
    const float* hs     = (const float*)d_in[0];
    const float* res    = (const float*)d_in[1];
    const float* norm_w = (const float*)d_in[2];
    const float* W_in   = (const float*)d_in[3];
    const float* conv_w = (const float*)d_in[4];
    const float* conv_b = (const float*)d_in[5];
    const float* W_xprj = (const float*)d_in[6];
    const float* W_dt   = (const float*)d_in[7];
    const float* b_dt   = (const float*)d_in[8];
    const float* A_log  = (const float*)d_in[9];
    const float* Dp     = (const float*)d_in[10];
    const float* W_out  = (const float*)d_in[11];

    float* out_h   = (float*)d_out;
    float* out_res = out_h + (size_t)ROWS * DM;

    // workspace carve-outs (256B aligned)
    char* cur = (char*)d_ws;
    auto take = [&](size_t bytes) -> void* {
        void* p = (void*)cur;
        cur += (bytes + 255) & ~(size_t)255;
        return p;
    };
    _Float16* hh    = (_Float16*)take((size_t)ROWS * DM * 2);      // normed h, f16
    _Float16* Winh  = (_Float16*)take((size_t)XZN * DM * 2);
    _Float16* Wxph  = (_Float16*)take((size_t)XPN * DI * 2);
    _Float16* Wdth  = (_Float16*)take((size_t)DI * DTR * 2);
    _Float16* Wouth = (_Float16*)take((size_t)DM * DI * 2);
    float*    xz    = (float*)take((size_t)ROWS * XZN * 4);
    _Float16* xh    = (_Float16*)take((size_t)ROWS * DI * 2);      // silu(conv(x)), f16
    float*    xf    = (float*)take((size_t)ROWS * DI * 4);         // silu(conv(x)), f32
    float*    xdbl  = (float*)take((size_t)ROWS * XPN * 4);
    _Float16* dtinh = (_Float16*)take((size_t)ROWS * DTR * 2);
    float*    dt    = (float*)take((size_t)ROWS * DI * 4);
    _Float16* yh    = (_Float16*)take((size_t)ROWS * DI * 2);

    // 1) weights -> f16
    cvt_f32_f16<<<2048, 256, 0, stream>>>(W_in,   Winh,  (size_t)XZN * DM);
    cvt_f32_f16<<<512,  256, 0, stream>>>(W_xprj, Wxph,  (size_t)XPN * DI);
    cvt_f32_f16<<<512,  256, 0, stream>>>(W_dt,   Wdth,  (size_t)DI * DTR);
    cvt_f32_f16<<<1024, 256, 0, stream>>>(W_out,  Wouth, (size_t)DM * DI);

    // 2) residual add + RMSNorm (residual -> d_out second half)
    addnorm_kernel<<<ROWS, 256, 0, stream>>>(hs, res, norm_w, out_res, hh);

    // 3) in-proj: xz[8192,4096] = hh * W_in^T   (K=1024)
    gemm_nt_wmma<4, 0><<<(ROWS / 16) * (XZN / 64) / 8, 256, 0, stream>>>(
        hh, Winh, xz, nullptr, ROWS, XZN, DM, DM, DM, XZN);

    // 4) causal depthwise conv + SiLU
    conv_silu_kernel<<<dim3(ROWS, DI / 256), 256, 0, stream>>>(xz, conv_w, conv_b, xf, xh);

    // 5) x-proj: xdbl[8192,96] = xh * W_xproj^T  (K=2048)
    gemm_nt_wmma<2, 0><<<(ROWS / 16) * (XPN / 32) / 8, 256, 0, stream>>>(
        xh, Wxph, xdbl, nullptr, ROWS, XPN, DI, DI, DI, XPN);

    // 6) slice dt-rank part -> f16
    cvt_slice_f16<<<1024, 256, 0, stream>>>(xdbl, dtinh, ROWS, DTR, XPN);

    // 7) dt-proj + softplus: dt[8192,2048] = softplus(dtinh * W_dt^T + b_dt)  (K=64)
    gemm_nt_wmma<4, 1><<<(ROWS / 16) * (DI / 64) / 8, 256, 0, stream>>>(
        dtinh, Wdth, dt, b_dt, ROWS, DI, DTR, DTR, DTR, DI);

    // 8) selective scan + D skip + SiLU(z) gate -> yh (f16)
    scan_kernel<<<(BATCH * DI * 16) / 256, 256, 0, stream>>>(dt, xf, xdbl, xz, A_log, Dp, yh);

    // 9) out-proj: out[8192,1024] = yh * W_out^T  (K=2048)
    gemm_nt_wmma<4, 0><<<(ROWS / 16) * (DM / 64) / 8, 256, 0, stream>>>(
        yh, Wouth, out_h, nullptr, ROWS, DM, DI, DI, DI, DM);

    (void)in_sizes; (void)n_in; (void)out_size; (void)ws_size;
}
